// MambaBlock_18777597018519
// MI455X (gfx1250) — matrix operand order repaired and checked by reading, not compiled, not run on hardware
//
#include <hip/hip_runtime.h>
#include <hip/hip_bf16.h>
#include <math.h>

#define D_MODEL 1024
#define D_STATE 64
#define D_CONV  4
#define D_INNER 2048
#define SEQ     2048
#define BATCH   2
#define M_TOT   (BATCH * SEQ)   // 4096 rows

typedef __attribute__((ext_vector_type(16))) __bf16        v16bf;
typedef __attribute__((ext_vector_type(8)))  float         v8f;
typedef __attribute__((ext_vector_type(8)))  unsigned int  v8u;
typedef __attribute__((ext_vector_type(4)))  unsigned int  u32x4;
typedef __attribute__((ext_vector_type(8)))  int           i32x8;
typedef __attribute__((ext_vector_type(4)))  int           i32x4;

static __device__ __forceinline__ unsigned short f32_to_bf16(float f) {
    unsigned int u = __builtin_bit_cast(unsigned int, f);
    u += 0x7FFFu + ((u >> 16) & 1u);          // round to nearest even
    return (unsigned short)(u >> 16);
}
static __device__ __forceinline__ float bf16_to_f32(unsigned short h) {
    unsigned int u = ((unsigned int)h) << 16;
    return __builtin_bit_cast(float, u);
}
static __device__ __forceinline__ float silu_f(float x) {
    return x / (1.0f + __expf(-x));
}

// ---------------------------------------------------------------------------
// Tensor Data Mover: 2D tile (tile_w x tile_h elements, 2B/elem) from global
// row-major tensor into LDS (packed row-major at lds_off).
// D# bitfields per CDNA5 ISA ch.10.8 (group0/group1 tables).
// ---------------------------------------------------------------------------
static __device__ __forceinline__ void tdm_load_2d_bf16(
        unsigned int lds_off, unsigned long long gaddr,
        unsigned int row_elems,     // tensor_dim0 (elements per row)
        unsigned int nrows,         // tensor_dim1 (total rows)
        unsigned int stride_elems,  // tensor_dim0_stride
        unsigned int tile_w, unsigned int tile_h) {
    u32x4 g0;
    g0[0] = 1u;                                   // count=1, user descriptor
    g0[1] = lds_off;                              // lds_addr (bytes)
    g0[2] = (unsigned int)gaddr;                  // global_addr[31:0]
    g0[3] = (unsigned int)(gaddr >> 32) | 0x80000000u;  // addr[56:32] | type=2
    i32x8 g1;
    g1[0] = (int)(1u << 16);                      // wg_mask=0, data_size=1 (2B)
    g1[1] = (int)((row_elems & 0xFFFFu) << 16);   // tensor_dim0[15:0]
    g1[2] = (int)((row_elems >> 16) | ((nrows & 0xFFFFu) << 16));
    g1[3] = (int)((nrows >> 16) | (tile_w << 16));        // tile_dim0
    g1[4] = (int)(tile_h & 0xFFFFu);                      // tile_dim1, tile_dim2=0
    g1[5] = (int)stride_elems;                            // dim0_stride[31:0]
    g1[6] = 0;                                            // stride hi / dim1_stride
    g1[7] = 0;
    i32x4 z4 = {0, 0, 0, 0};
#if __clang_major__ >= 23
    i32x8 z8 = {0, 0, 0, 0, 0, 0, 0, 0};
    __builtin_amdgcn_tensor_load_to_lds(g0, g1, z4, z4, z8, 0);
#else
    __builtin_amdgcn_tensor_load_to_lds(g0, g1, z4, z4, 0);
#endif
}

// ---------------------------------------------------------------------------
// fp32 -> bf16 conversion
// ---------------------------------------------------------------------------
__global__ void cvt_f32_bf16(const float* __restrict__ in,
                             unsigned short* __restrict__ out, int n) {
    int i = blockIdx.x * blockDim.x + threadIdx.x;
    if (i < n) out[i] = f32_to_bf16(in[i]);
}

// ---------------------------------------------------------------------------
// bf16 WMMA GEMM with TDM->LDS staging:
//   out(M,N) = A(M,K) x W(N,K)^T, fp32 accumulate.
// Block = 256 threads = 8 waves (2 M x 4 N wave grid) computing a 32x64 tile.
// Per 32-wide k-step: wave 0 DMAs a 32x32 A tile + 64x32 B tile into LDS
// (tensor_load_to_lds, s_wait_tensorcnt), then each wave feeds one
// v_wmma_f32_16x16x32_bf16 from ds_load_b128 fragments.
// EPI 0: store fp32 ; EPI 1: fp32 + bias ;
// EPI 2: Mamba epilogue bf16( (acc + xconv*D) * silu(gate) )
// ---------------------------------------------------------------------------
template <int EPI>
__global__ void gemm_wmma_bf16(const unsigned short* __restrict__ Abf, int lda,
                               const unsigned short* __restrict__ Wbf, int ldb,
                               int K, int Nrows,
                               float* __restrict__ outF, int ldc,
                               const float* __restrict__ bias,
                               const unsigned short* __restrict__ xconvbf,
                               const float* __restrict__ Dvec,
                               const float* __restrict__ xproj,   // gate src
                               unsigned short* __restrict__ outBf) {
    extern __shared__ unsigned short smem[];   // [0,1024): A 32x32, [1024,3072): B 64x32
    unsigned short* lds_a = smem;
    unsigned short* lds_b = smem + 1024;

    const int lane = threadIdx.x & 31;
    const int wave = threadIdx.x >> 5;
    const int g    = lane >> 4;     // half-wave select
    const int ln   = lane & 15;
    const int mrow = (wave >> 2) * 16 + ln;   // 0..31 row inside A tile
    const int nrow = (wave & 3) * 16 + ln;    // 0..63 row inside B tile

    const unsigned long long baseA = (unsigned long long)(uintptr_t)Abf +
                                     (unsigned long long)(blockIdx.x * 32) * lda * 2ull;
    const unsigned long long baseB = (unsigned long long)(uintptr_t)Wbf +
                                     (unsigned long long)(blockIdx.y * 64) * ldb * 2ull;

    v8f acc = {};

    for (int k0 = 0; k0 < K; k0 += 32) {
        if (wave == 0) {
            tdm_load_2d_bf16(0u,    baseA + 2ull * k0, (unsigned)lda, M_TOT,
                             (unsigned)lda, 32u, 32u);
            tdm_load_2d_bf16(2048u, baseB + 2ull * k0, (unsigned)ldb, (unsigned)Nrows,
                             (unsigned)ldb, 32u, 64u);
            __builtin_amdgcn_s_wait_tensorcnt(0);
        }
        __syncthreads();

        // A 16x32 bf16 fragment (ISA 7.12.2): lanes 0-15 K={0..7,16..23},
        // lanes 16-31 K={8..15,24..31} -> two 16B chunks per lane.
        uint4 a0 = *(const uint4*)(lds_a + mrow * 32 + 8 * g);
        uint4 a1 = *(const uint4*)(lds_a + mrow * 32 + 16 + 8 * g);
        // B 32x16 fragment: lane = column n; lanes 0-15 K={0..7,16..23}, 16-31 K={8..15,24..31}.
        uint4 b0 = *(const uint4*)(lds_b + nrow * 32 + 8 * g);
        uint4 b1 = *(const uint4*)(lds_b + nrow * 32 + 16 + 8 * g);

        v8u au = {a0.x, a0.y, a0.z, a0.w, a1.x, a1.y, a1.z, a1.w};
        v8u bu = {b0.x, b0.y, b0.z, b0.w, b1.x, b1.y, b1.z, b1.w};
        v16bf a = __builtin_bit_cast(v16bf, au);
        v16bf b = __builtin_bit_cast(v16bf, bu);
        acc = __builtin_amdgcn_wmma_f32_16x16x32_bf16(
                  false, a, false, b, (short)0, acc, false, false);

        __syncthreads();   // protect LDS tiles before next DMA overwrites
    }

    const int m0 = blockIdx.x * 32 + (wave >> 2) * 16;
    const int n0 = blockIdx.y * 64 + (wave & 3) * 16;
#pragma unroll
    for (int r = 0; r < 8; ++r) {
        // C/D layout: VGPR r -> M=r (lanes 0-15), M=r+8 (lanes 16-31); N=lane&15
        int m = m0 + r + 8 * g;
        int n = n0 + ln;
        float v = acc[r];
        if (EPI == 0) {
            outF[(size_t)m * ldc + n] = v;
        } else if (EPI == 1) {
            outF[(size_t)m * ldc + n] = v + bias[n];
        } else {
            float xc   = bf16_to_f32(xconvbf[(size_t)m * D_INNER + n]);
            float gate = xproj[(size_t)m * (2 * D_INNER) + D_INNER + n];
            float val  = (v + xc * Dvec[n]) * silu_f(gate);
            outBf[(size_t)m * D_INNER + n] = f32_to_bf16(val);
        }
    }
}

// ---------------------------------------------------------------------------
// Depthwise causal conv1d (k=4, left pad 3) + SiLU; reads x_ssm half of xproj.
// ---------------------------------------------------------------------------
__global__ void conv_silu_kernel(const float* __restrict__ xproj,
                                 const float* __restrict__ conv_w,
                                 const float* __restrict__ conv_b,
                                 unsigned short* __restrict__ xconvbf) {
    int idx = blockIdx.x * blockDim.x + threadIdx.x;
    if (idx >= M_TOT * D_INNER) return;
    int c   = idx % D_INNER;
    int row = idx / D_INNER;
    int t   = row % SEQ;
    float acc = conv_b[c];
#pragma unroll
    for (int j = 0; j < D_CONV; ++j) {
        int tt = t - (D_CONV - 1) + j;
        if (tt >= 0)
            acc += conv_w[c * D_CONV + j] *
                   xproj[(size_t)(row - (D_CONV - 1) + j) * (2 * D_INNER) + c];
    }
    xconvbf[idx] = f32_to_bf16(silu_f(acc));
}

// ---------------------------------------------------------------------------
// Sequential part of the scan, with x@B^T hoisted out: per time step only
// h = tanh(h @ A^T + xB_t).  One 64-thread block per batch; A rows in
// registers, h broadcast from LDS. Post-update h stored as bf16 for h@C^T.
// ---------------------------------------------------------------------------
__global__ void scan_kernel(const float* __restrict__ Amat,
                            const float* __restrict__ xB,
                            unsigned short* __restrict__ hsbf) {
    __shared__ float h[D_STATE];
    int i = threadIdx.x;                 // state index 0..63
    int b = blockIdx.x;
    float areg[D_STATE];
#pragma unroll
    for (int j = 0; j < D_STATE; ++j) areg[j] = Amat[i * D_STATE + j];
    h[i] = 0.0f;
    __syncthreads();
    size_t base = (size_t)b * SEQ;
    for (int t = 0; t < SEQ; ++t) {
        float acc = xB[(base + t) * D_STATE + i];
#pragma unroll 16
        for (int j = 0; j < D_STATE; ++j) acc += h[j] * areg[j];
        float hn = tanhf(acc);
        __syncthreads();
        h[i] = hn;
        hsbf[(base + t) * D_STATE + i] = f32_to_bf16(hn);
        __syncthreads();
    }
}

// ---------------------------------------------------------------------------
extern "C" void kernel_launch(void* const* d_in, const int* in_sizes, int n_in,
                              void* d_out, int out_size, void* d_ws, size_t ws_size,
                              hipStream_t stream) {
    const float* x  = (const float*)d_in[0];
    const float* w1 = (const float*)d_in[1];
    const float* b1 = (const float*)d_in[2];
    const float* cw = (const float*)d_in[3];
    const float* cb = (const float*)d_in[4];
    const float* Am = (const float*)d_in[5];
    const float* Bm = (const float*)d_in[6];
    const float* Cm = (const float*)d_in[7];
    const float* Dv = (const float*)d_in[8];
    const float* w2 = (const float*)d_in[9];
    const float* b2 = (const float*)d_in[10];
    float* out = (float*)d_out;

    char* ws = (char*)d_ws;
    size_t off = 0;
    auto alloc = [&](size_t bytes) -> void* {
        void* p = ws + off;
        off = (off + bytes + 255) & ~(size_t)255;
        return p;
    };
    unsigned short* xbf     = (unsigned short*)alloc((size_t)M_TOT * D_MODEL * 2);
    unsigned short* w1bf    = (unsigned short*)alloc((size_t)2 * D_INNER * D_MODEL * 2);
    float*          xproj   = (float*)alloc((size_t)M_TOT * 2 * D_INNER * 4);
    unsigned short* xconvbf = (unsigned short*)alloc((size_t)M_TOT * D_INNER * 2);
    unsigned short* Bbf     = (unsigned short*)alloc((size_t)D_STATE * D_INNER * 2);
    float*          xB      = (float*)alloc((size_t)M_TOT * D_STATE * 4);
    unsigned short* hsbf    = (unsigned short*)alloc((size_t)M_TOT * D_STATE * 2);
    unsigned short* Cbf     = (unsigned short*)alloc((size_t)D_INNER * D_STATE * 2);
    unsigned short* xoutbf  = (unsigned short*)alloc((size_t)M_TOT * D_INNER * 2);
    unsigned short* w2bf    = (unsigned short*)alloc((size_t)D_MODEL * D_INNER * 2);

    auto cvt = [&](const float* src, unsigned short* dst, int n) {
        cvt_f32_bf16<<<(n + 255) / 256, 256, 0, stream>>>(src, dst, n);
    };
    cvt(x,  xbf,  M_TOT * D_MODEL);
    cvt(w1, w1bf, 2 * D_INNER * D_MODEL);
    cvt(Bm, Bbf,  D_STATE * D_INNER);
    cvt(Cm, Cbf,  D_INNER * D_STATE);
    cvt(w2, w2bf, D_MODEL * D_INNER);

    const size_t shmem = (32 * 32 + 64 * 32) * sizeof(unsigned short);  // 6 KB

    // GEMM1: xproj = x @ in_proj_w^T + b   (4096 x 4096 x 1024)
    gemm_wmma_bf16<1><<<dim3(M_TOT / 32, (2 * D_INNER) / 64), 256, shmem, stream>>>(
        xbf, D_MODEL, w1bf, D_MODEL, D_MODEL, 2 * D_INNER, xproj, 2 * D_INNER, b1,
        nullptr, nullptr, nullptr, nullptr);

    // depthwise conv + SiLU -> bf16
    conv_silu_kernel<<<(M_TOT * D_INNER + 255) / 256, 256, 0, stream>>>(
        xproj, cw, cb, xconvbf);

    // GEMM2: xB = x_conv @ B^T   (4096 x 64 x 2048) — hoisted scan input
    gemm_wmma_bf16<0><<<dim3(M_TOT / 32, D_STATE / 64), 256, shmem, stream>>>(
        xconvbf, D_INNER, Bbf, D_INNER, D_INNER, D_STATE, xB, D_STATE, nullptr,
        nullptr, nullptr, nullptr, nullptr);

    // sequential 64-state recurrence (tiny)
    scan_kernel<<<BATCH, D_STATE, 0, stream>>>(Am, xB, hsbf);

    // GEMM3 fused: xout = (hs @ C^T + x_conv*D) * silu(gate)  -> bf16
    gemm_wmma_bf16<2><<<dim3(M_TOT / 32, D_INNER / 64), 256, shmem, stream>>>(
        hsbf, D_STATE, Cbf, D_STATE, D_STATE, D_INNER, nullptr, 0, nullptr,
        xconvbf, Dv, xproj, xoutbf);

    // GEMM4: out = xout @ out_proj_w^T + b   (4096 x 1024 x 2048)
    gemm_wmma_bf16<1><<<dim3(M_TOT / 32, D_MODEL / 64), 256, shmem, stream>>>(
        xoutbf, D_INNER, w2bf, D_INNER, D_INNER, D_MODEL, out, D_MODEL, b2,
        nullptr, nullptr, nullptr, nullptr);
}
